// MMFConv2d_45672682226145
// MI455X (gfx1250) — compile-verified, matrix operand order checked by reading, not run on hardware
//
#include <hip/hip_runtime.h>
#include <hip/hip_bf16.h>

typedef __attribute__((ext_vector_type(16))) __bf16 v16bf;
typedef __attribute__((ext_vector_type(8)))  __bf16 v8bf;
typedef __attribute__((ext_vector_type(8)))  float  v8f;

#define CIN   32
#define COUT  64
#define HDIM  256
#define WDIM  256
#define TW    64     // output columns per workgroup
#define NPOS  68     // padded staging positions (66 needed: TW + 2 halo)
#define KPAD  40     // padded channel stride in LDS elements (80B -> conflict-free b128)

// ---------------------------------------------------------------------------
// Prep: ternarize weights (STE sign) -> bf16, stored in WMMA A-fragment order:
//   wsA[((wtile*9 + tap)*32 + lane) * 16 + hh]
// A layout (16-bit 16x32): lane m = lane&15; half hh -> k = (hh>=8?16:0)+(lane>=16?8:0)+(hh&7)
// Total 4*9*32*16 bf16 = 36 KB in d_ws. Runs once per launch; L2-resident after.
// ---------------------------------------------------------------------------
__global__ void prep_weights_kernel(const float* __restrict__ wt,
                                    __bf16* __restrict__ wsA)
{
    const int idx = blockIdx.x * blockDim.x + threadIdx.x;   // (wtile*9+t)*32+lane
    if (idx >= 4 * 9 * 32) return;
    const int lane  = idx & 31;
    const int t     = (idx >> 5) % 9;
    const int wtile = idx / (9 * 32);
    const int kh = t / 3, kw = t % 3;
    const int mrow = lane & 15;
    const int koff = (lane >> 4) * 8;
    const int co   = wtile * 16 + mrow;

    v16bf frag;
#pragma unroll
    for (int hh = 0; hh < 16; ++hh) {
        int k = ((hh >= 8) ? 16 : 0) + koff + (hh & 7);
        float wv = wt[(((co) * CIN + k) * 3 + kh) * 3 + kw];
        float ws = (wv > 0.f) ? 1.f : ((wv < 0.f) ? -1.f : 0.f);  // STE sign()
        frag[hh] = (__bf16)ws;
    }
    *(v16bf*)&wsA[idx * 16] = frag;
}

// ---------------------------------------------------------------------------
// Main: implicit-GEMM 3x3 conv via bf16 WMMA with hi/lo input split.
// Dual accumulator chains + explicit double-buffered B-fragment loads so the
// LDS pipe stays ahead of the WMMA pipe (no full dscnt drains).
// ---------------------------------------------------------------------------
__launch_bounds__(128)
__global__ void ternary_conv3x3_wmma(const float* __restrict__ x,
                                     const __bf16* __restrict__ wsA,
                                     const float* __restrict__ bias,
                                     const float* __restrict__ scale,
                                     float* __restrict__ out)
{
    __shared__ __bf16 sHi[3 * NPOS * KPAD];
    __shared__ __bf16 sLo[3 * NPOS * KPAD];

    const int tid  = threadIdx.x;
    const int lane = tid & 31;
    const int wave = tid >> 5;            // 0..3 -> C_out tile of 16
    const int blk  = blockIdx.x;

    const int tile = blk & 3;             // W strip (4 strips of 64)
    const int h    = (blk >> 2) & 255;    // output row
    const int b    = blk >> 10;           // batch
    const int w0   = tile * TW;

    // ---- A fragments: one 32B coalesced load per tap (prepped layout) ----
    v16bf aFrag[9];
#pragma unroll
    for (int t = 0; t < 9; ++t)
        aFrag[t] = *(const v16bf*)&wsA[((wave * 9 + t) * 32 + lane) * 16];

    // ---- Stage 3 input rows into LDS as bf16 hi/lo split ----
    // x = hi + lo with hi = bf16(x), lo = bf16(x - hi)  (~16 mantissa bits kept)
    for (int e = tid; e < 3 * CIN * 66; e += 128) {
        int p   = e % 66;
        int t   = e / 66;
        int ci  = t % CIN;
        int r   = t / CIN;
        int win = w0 + p - 1;
        int hin = h + r - 1;
        float v = 0.0f;
        if ((unsigned)win < (unsigned)WDIM && (unsigned)hin < (unsigned)HDIM)
            v = x[(((b * CIN + ci) * HDIM + hin) * WDIM) + win];
        __bf16 hi = (__bf16)v;
        __bf16 lo = (__bf16)(v - (float)hi);
        int di = (r * NPOS + p) * KPAD + ci;
        sHi[di] = hi;
        sLo[di] = lo;
    }

    __syncthreads();

    const float sc   = scale[0];
    const int   mhi  = (lane >> 4) * 8;   // C/D: reg r -> M = r + (lane>=16 ? 8 : 0)
    const int   co0  = wave * 16;
    const int   koff = (lane >> 4) * 8;
    float bv[8];
#pragma unroll
    for (int r = 0; r < 8; ++r)
        bv[r] = bias[co0 + r + mhi];

    // ---- 4 x 16-wide output tiles; 9 taps; hi+lo WMMA per tap ----
    // B layout (16-bit 32x16): lane n = lane&15; halves 0..7 -> k = koff+0..7,
    // halves 8..15 -> k = 16+koff+0..7. Two contiguous 16B runs in LDS.
    const int ncol = lane & 15;
#pragma unroll
    for (int nt = 0; nt < 4; ++nt) {
        v8f accH = {};
        v8f accL = {};

        // prologue: tap 0 fragment
        int bidx = (0 * NPOS + nt * 16 + ncol + 0) * KPAD + koff;
        v8bf h0 = *(const v8bf*)&sHi[bidx];
        v8bf h1 = *(const v8bf*)&sHi[bidx + 16];
        v8bf l0 = *(const v8bf*)&sLo[bidx];
        v8bf l1 = *(const v8bf*)&sLo[bidx + 16];

#pragma unroll
        for (int t = 0; t < 9; ++t) {
            // issue next tap's loads before this tap's WMMAs (double buffer)
            v8bf nh0 = h0, nh1 = h1, nl0 = l0, nl1 = l1;
            if (t < 8) {
                const int t2   = t + 1;
                const int kh2  = t2 / 3, kw2 = t2 % 3;
                const int nidx = (kh2 * NPOS + nt * 16 + ncol + kw2) * KPAD + koff;
                nh0 = *(const v8bf*)&sHi[nidx];
                nh1 = *(const v8bf*)&sHi[nidx + 16];
                nl0 = *(const v8bf*)&sLo[nidx];
                nl1 = *(const v8bf*)&sLo[nidx + 16];
            }
            v16bf bHi = __builtin_shufflevector(h0, h1, 0,1,2,3,4,5,6,7,8,9,10,11,12,13,14,15);
            v16bf bLo = __builtin_shufflevector(l0, l1, 0,1,2,3,4,5,6,7,8,9,10,11,12,13,14,15);
            accH = __builtin_amdgcn_wmma_f32_16x16x32_bf16(false, aFrag[t], false, bHi,
                                                           (short)0, accH, false, false);
            accL = __builtin_amdgcn_wmma_f32_16x16x32_bf16(false, aFrag[t], false, bLo,
                                                           (short)0, accL, false, false);
            h0 = nh0; h1 = nh1; l0 = nl0; l1 = nl1;
        }

        // epilogue: merge chains, scale, bias, clamp, store
        const int wout = w0 + nt * 16 + ncol;
#pragma unroll
        for (int r = 0; r < 8; ++r) {
            const int co = co0 + r + mhi;
            float v = sc * (accH[r] + accL[r]) + bv[r];
            v = fminf(fmaxf(v, -10000.f), 10000.f);
            out[(((b * COUT + co) * HDIM + h) * WDIM) + wout] = v;
        }
    }
}

extern "C" void kernel_launch(void* const* d_in, const int* in_sizes, int n_in,
                              void* d_out, int out_size, void* d_ws, size_t ws_size,
                              hipStream_t stream) {
    (void)in_sizes; (void)n_in; (void)out_size; (void)ws_size;
    const float* x     = (const float*)d_in[0];
    const float* wt    = (const float*)d_in[1];
    const float* bias  = (const float*)d_in[2];
    const float* scale = (const float*)d_in[3];
    float*       out   = (float*)d_out;
    __bf16*      wsA   = (__bf16*)d_ws;   // 36 KB of A-fragment bf16 weights

    // One-time (per launch) weight ternarize + bf16 fragment pack.
    prep_weights_kernel<<<9, 128, 0, stream>>>(wt, wsA);

    // grid: 4 W-strips x 256 rows x 16 batches = 16384 workgroups of 128 threads
    dim3 grid(4 * HDIM * 16);
    dim3 block(128);
    ternary_conv3x3_wmma<<<grid, block, 0, stream>>>(x, wsA, bias, scale, out);
}
